// BiLSTMCRF_34282428956700
// MI455X (gfx1250) — matrix operand order repaired and checked by reading, not compiled, not run on hardware
//
#include <hip/hip_runtime.h>
#include <hip/hip_bf16.h>

// ---------------------------------------------------------------------------
// BiLSTM-CRF loss for MI455X (gfx1250, wave32, WMMA).
//
//  * bilstm_scan: 2 persistent workgroups (1 per direction), 512 thr = 16
//    waves (4 waves/SIMD -> ~256 VGPR/wave budget). Per timestep one fused
//    GEMM  g = [h|x](64x384) @ Wcomb^T(384x1024) via v_wmma_f32_16x16x32_f16.
//    Wave w owns M-tile (w&3) and h-column group (w>>2)*64 with all 4 gates:
//    acc[gate][j] (128 VGPRs), cell state c_st[j] (32 VGPRs) resident for all
//    512 steps; h in LDS. W loads stay inside the time loop via an opaque-
//    pointer asm barrier (stops LICM hoist -> scratch spill), and are issued
//    through an explicit address_space(1) pointer so they lower to
//    global_load_b128 (LOADcnt only) instead of flat_load (LOADcnt+DScnt).
//    W_comb (768 KB/dir) is L2/WGP$-resident: re-reading per step is the
//    right data path on this chip (192 MB L2).
//  * fc_gemm: (32768x512)@(512x64pad) WMMA GEMM, 1 wave / 16-row tile.
//  * log_softmax50 + crf_loss: CRF forward recursion in one workgroup.
//
// Workspace layout (~44 MB):
//   [0)          Wcomb_f  f16 1024x384   (W_hh_f | W_ih_f)
//   [786432)     Wcomb_b  f16 1024x384
//   [1572864)    fcW_pad  f16 64x512     (rows 50..63 zero)
//   [1638400)    h_cat    f16 32768x512  ([h_f | h_b], time-unreversed)
//   [35192832)   feats    f32 32768x64   (logits in cols 0..49 after softmax)
// ---------------------------------------------------------------------------

typedef _Float16 v8h  __attribute__((ext_vector_type(8)));
typedef _Float16 v16h __attribute__((ext_vector_type(16)));
typedef float    v8f  __attribute__((ext_vector_type(8)));

// Global-address-space pointer to a 16 x f16 fragment (forces global_load).
typedef const v16h __attribute__((address_space(1)))* gv16p;

#define BB 64
#define TT 512
#define DD 128
#define HH 256
#define G4 1024   // 4*H
#define KC 384    // H + D
#define CC 50
#define CP 64     // padded classes

// A-fragment for v_wmma_f32_16x16x32_f16: lane holds row (lane%16);
// elements 0..15 = K[16*half .. +16)  (half=lane/16).
__device__ __forceinline__ v16h frag_a(const _Float16* p_lo) {
  v8h lo = *(const v8h*)(p_lo);
  v8h hi = *(const v8h*)(p_lo + 8);
  v16h r;
#pragma unroll
  for (int i = 0; i < 8; ++i) { r[i] = lo[i]; r[i + 8] = hi[i]; }
  return r;
}

__device__ __forceinline__ float sigf(float x) {
  return 1.0f / (1.0f + __expf(-x));
}

// --------------------------- weight prep -----------------------------------
__global__ void prep_wcomb(const float* __restrict__ W_ih,
                           const float* __restrict__ W_hh,
                           _Float16* __restrict__ dst) {
  int i = blockIdx.x * blockDim.x + threadIdx.x;
  if (i >= G4 * KC) return;
  int n = i / KC, k = i % KC;
  float v = (k < HH) ? W_hh[n * HH + k] : W_ih[n * DD + (k - HH)];
  dst[i] = (_Float16)v;
}

__global__ void prep_fcw(const float* __restrict__ fcW,
                         _Float16* __restrict__ dst) {
  int i = blockIdx.x * blockDim.x + threadIdx.x;
  if (i >= CP * 2 * HH) return;
  int n = i / (2 * HH), k = i % (2 * HH);
  dst[i] = (_Float16)((n < CC) ? fcW[n * 2 * HH + k] : 0.0f);
}

// --------------------------- fused BiLSTM scan ------------------------------
__global__ __launch_bounds__(512, 1) void bilstm_scan(
    const int* __restrict__ words, const int* __restrict__ seq_len,
    const float* __restrict__ emb,
    const _Float16* __restrict__ wc_f, const _Float16* __restrict__ wc_b,
    const float* __restrict__ b_f, const float* __restrict__ b_b,
    _Float16* __restrict__ h_cat) {
  const int dir = blockIdx.x;                     // 0 = fwd, 1 = bwd
  const _Float16* __restrict__ Wc = dir ? wc_b : wc_f;
  const float* __restrict__ bias = dir ? b_b : b_f;

  __shared__ _Float16 h_buf[BB * HH];             // 32 KB
  __shared__ _Float16 x_buf[BB * DD];             // 16 KB
  __shared__ float bias_s[G4];                    // 4 KB
  __shared__ int Ls[BB];

  const int tid  = threadIdx.x;
  const int lane = tid & 31;
  const int wv   = tid >> 5;                      // 0..15
  const int mt   = wv & 3;                        // M-tile (16 batch rows)
  const int hg   = wv >> 2;                       // 0..3: 64 h-cols each
  const int lrow = lane & 15;
  const int half = lane >> 4;

  for (int i = tid; i < BB * HH; i += 512) h_buf[i] = (_Float16)0.0f;
  for (int i = tid; i < G4; i += 512) bias_s[i] = bias[i];
  if (tid < BB) Ls[tid] = seq_len[tid];

  // Cell state in D-fragment layout, resident in VGPRs for all 512 steps.
  v8f c_st[4];
#pragma unroll
  for (int j = 0; j < 4; ++j)
#pragma unroll
    for (int r = 0; r < 8; ++r) c_st[j][r] = 0.0f;

  __syncthreads();

  const int xrow = tid >> 3;                      // 0..63
  const int xcol = (tid & 7) * 16;

#pragma unroll 1
  for (int t = 0; t < TT; ++t) {
    // Gather embedding rows for this step (time-reversed for backward dir).
    {
      int L = Ls[xrow];
      int ts = dir ? ((t < L) ? (L - 1 - t) : t) : t;
      int w = words[xrow * TT + ts];
      const float* src = emb + (long)w * DD + xcol;
      _Float16* dst = x_buf + xrow * DD + xcol;
#pragma unroll
      for (int i = 0; i < 16; ++i) dst[i] = (_Float16)src[i];
    }
    __syncthreads();

    // acc = bias (read per (g,j) from LDS, broadcast over the 8 rows)
    v8f acc[4][4];
#pragma unroll
    for (int g = 0; g < 4; ++g) {
#pragma unroll
      for (int j = 0; j < 4; ++j) {
        float bv = bias_s[g * HH + hg * 64 + j * 16 + lrow];
#pragma unroll
        for (int r = 0; r < 8; ++r) acc[g][j][r] = bv;
      }
    }

    // g += [h | x] @ Wcomb^T  : K = 384 = 12 k-steps of 32
#pragma unroll 1
    for (int ks = 0; ks < 12; ++ks) {
      v16h a;
      if (ks < 8) {
        a = frag_a(h_buf + (mt * 16 + lrow) * HH + ks * 32 + half * 16);
      } else {
        a = frag_a(x_buf + (mt * 16 + lrow) * DD + (ks - 8) * 32 + half * 16);
      }
#pragma unroll
      for (int g = 0; g < 4; ++g) {
        // Opaque base pointer: zero-instruction barrier that stops LICM from
        // hoisting the (t-invariant) weight loads out of the time loop, and
        // bounds how many B fragments the scheduler keeps in flight.
        unsigned long long wp = (unsigned long long)Wc;
        asm volatile("" : "+s"(wp));
#pragma unroll
        for (int j = 0; j < 4; ++j) {
          int n = g * HH + hg * 64 + j * 16 + lrow;  // gate column
          unsigned long long addr =
              wp + 2ull * ((unsigned long long)n * KC + ks * 32 + half * 16);
          v16h bf = *(gv16p)addr;   // address_space(1) -> global_load_b128
          acc[g][j] = __builtin_amdgcn_wmma_f32_16x16x32_f16(
              false, a, false, bf, (short)0, acc[g][j], false, false);
        }
      }
    }
    __syncthreads();  // all LDS reads done before h_buf is rewritten

    // Cell update (in-register) + h writeback to LDS and global.
#pragma unroll
    for (int j = 0; j < 4; ++j) {
#pragma unroll
      for (int r = 0; r < 8; ++r) {
        float iv = sigf(acc[0][j][r]);
        float fv = sigf(acc[1][j][r]);
        float gv = tanhf(acc[2][j][r]);
        float ov = sigf(acc[3][j][r]);
        float c = fv * c_st[j][r] + iv * gv;
        float h = ov * tanhf(c);
        c_st[j][r] = c;
        int row = mt * 16 + half * 8 + r;            // batch index
        int col = hg * 64 + j * 16 + lrow;           // h index
        _Float16 hh = (_Float16)h;
        h_buf[row * HH + col] = hh;
        int L = Ls[row];
        int to = dir ? ((t < L) ? (L - 1 - t) : t) : t;  // un-reverse output
        h_cat[((long)row * TT + to) * (2 * HH) + dir * HH + col] = hh;
      }
    }
    __syncthreads();
  }
}

// --------------------------- fc layer GEMM ----------------------------------
__global__ __launch_bounds__(256) void fc_gemm(
    const _Float16* __restrict__ h_cat, const _Float16* __restrict__ fcw,
    const float* __restrict__ fc_b, float* __restrict__ feats) {
  const int lane = threadIdx.x & 31;
  const int mt = (blockIdx.x * blockDim.x + threadIdx.x) >> 5;  // M-tile id
  const int lrow = lane & 15, half = lane >> 4;

  v8f acc[4];
#pragma unroll
  for (int nt = 0; nt < 4; ++nt)
#pragma unroll
    for (int r = 0; r < 8; ++r) acc[nt][r] = 0.0f;

#pragma unroll 1
  for (int ks = 0; ks < 16; ++ks) {  // K = 512
    v16h a = frag_a(h_cat + ((long)(mt * 16 + lrow)) * (2 * HH) + ks * 32 +
                    half * 16);
#pragma unroll
    for (int nt = 0; nt < 4; ++nt) {
      v16h bf = *(const v16h*)(fcw + (long)(nt * 16 + lrow) * (2 * HH) +
                               ks * 32 + half * 16);
      acc[nt] = __builtin_amdgcn_wmma_f32_16x16x32_f16(
          false, a, false, bf, (short)0, acc[nt], false, false);
    }
  }
#pragma unroll
  for (int nt = 0; nt < 4; ++nt) {
#pragma unroll
    for (int r = 0; r < 8; ++r) {
      int row = mt * 16 + half * 8 + r;
      int n = nt * 16 + lrow;
      feats[(long)row * CP + n] = acc[nt][r] + ((n < CC) ? fc_b[n] : 0.0f);
    }
  }
}

// --------------------------- log_softmax over C=50 --------------------------
__global__ void log_softmax50(float* __restrict__ feats) {
  int r = blockIdx.x * blockDim.x + threadIdx.x;
  if (r >= BB * TT) return;
  float* p = feats + (long)r * CP;
  float m = -3.4e38f;
  for (int c = 0; c < CC; ++c) m = fmaxf(m, p[c]);
  float s = 0.0f;
  for (int c = 0; c < CC; ++c) s += __expf(p[c] - m);
  float ls = m + __logf(s);
  for (int c = 0; c < CC; ++c) p[c] -= ls;
}

// --------------------------- CRF forward + loss -----------------------------
__global__ __launch_bounds__(1024) void crf_loss(
    const float* __restrict__ logits, const int* __restrict__ target,
    const int* __restrict__ seq_len, const float* __restrict__ trans,
    const float* __restrict__ start_s, const float* __restrict__ end_s,
    float* __restrict__ out) {
  __shared__ float al[2][BB][CC];
  __shared__ float tr[CC][CC];
  __shared__ float red[BB];
  const int tid = threadIdx.x;

  for (int i = tid; i < CC * CC; i += 1024) tr[i / CC][i % CC] = trans[i];
  for (int p = tid; p < BB * CC; p += 1024) {
    int b = p / CC, c = p % CC;
    al[0][b][c] = logits[((long)b * TT) * CP + c] + start_s[c];
  }
  __syncthreads();

  int cur = 0;
#pragma unroll 1
  for (int t = 1; t < TT; ++t) {
    int nxt = cur ^ 1;
    for (int p = tid; p < BB * CC; p += 1024) {
      int b = p / CC, c2 = p % CC;
      float v;
      if (t < seq_len[b]) {
        float m = -3.4e38f;
        for (int c = 0; c < CC; ++c) m = fmaxf(m, al[cur][b][c] + tr[c][c2]);
        float s = 0.0f;
        for (int c = 0; c < CC; ++c)
          s += __expf(al[cur][b][c] + tr[c][c2] - m);
        v = m + __logf(s) + logits[((long)b * TT + t) * CP + c2];
      } else {
        v = al[cur][b][c2];
      }
      al[nxt][b][c2] = v;
    }
    __syncthreads();
    cur = nxt;
  }

  if (tid < BB) {
    int b = tid;
    float m = -3.4e38f;
    for (int c = 0; c < CC; ++c) m = fmaxf(m, al[cur][b][c] + end_s[c]);
    float s = 0.0f;
    for (int c = 0; c < CC; ++c) s += __expf(al[cur][b][c] + end_s[c] - m);
    float norm = m + __logf(s);

    int L = seq_len[b];
    const int* tg = target + b * TT;
    float gold = start_s[tg[0]] + end_s[tg[L - 1]];
    for (int t = 0; t < TT; ++t) {
      if (t < L) {
        gold += logits[((long)b * TT + t) * CP + tg[t]];
        if (t >= 1) gold += tr[tg[t - 1]][tg[t]];
      }
    }
    red[b] = norm - gold;
  }
  __syncthreads();
  if (tid == 0) {
    float s = 0.0f;
    for (int b = 0; b < BB; ++b) s += red[b];
    out[0] = s / (float)BB;
  }
}

// --------------------------- launcher ---------------------------------------
extern "C" void kernel_launch(void* const* d_in, const int* in_sizes, int n_in,
                              void* d_out, int out_size, void* d_ws,
                              size_t ws_size, hipStream_t stream) {
  const int*   words   = (const int*)d_in[0];
  const int*   target  = (const int*)d_in[1];
  const int*   seq_len = (const int*)d_in[2];
  const float* emb     = (const float*)d_in[3];
  const float* W_ih_f  = (const float*)d_in[4];
  const float* W_hh_f  = (const float*)d_in[5];
  const float* b_f     = (const float*)d_in[6];
  const float* W_ih_b  = (const float*)d_in[7];
  const float* W_hh_b  = (const float*)d_in[8];
  const float* b_b     = (const float*)d_in[9];
  const float* fc_W    = (const float*)d_in[10];
  const float* fc_b    = (const float*)d_in[11];
  const float* trans   = (const float*)d_in[12];
  const float* start_s = (const float*)d_in[13];
  const float* end_s   = (const float*)d_in[14];

  char* ws = (char*)d_ws;
  _Float16* wc_f  = (_Float16*)(ws + 0);
  _Float16* wc_b  = (_Float16*)(ws + 786432);
  _Float16* fcw   = (_Float16*)(ws + 1572864);
  _Float16* h_cat = (_Float16*)(ws + 1638400);
  float*    feats = (float*)(ws + 35192832);

  hipLaunchKernelGGL(prep_wcomb, dim3((G4 * KC + 255) / 256), dim3(256), 0,
                     stream, W_ih_f, W_hh_f, wc_f);
  hipLaunchKernelGGL(prep_wcomb, dim3((G4 * KC + 255) / 256), dim3(256), 0,
                     stream, W_ih_b, W_hh_b, wc_b);
  hipLaunchKernelGGL(prep_fcw, dim3((CP * 2 * HH + 255) / 256), dim3(256), 0,
                     stream, fc_W, fcw);
  hipLaunchKernelGGL(bilstm_scan, dim3(2), dim3(512), 0, stream, words,
                     seq_len, emb, wc_f, wc_b, b_f, b_b, h_cat);
  hipLaunchKernelGGL(fc_gemm, dim3((BB * TT / 16) / 8), dim3(256), 0, stream,
                     h_cat, fcw, fc_b, feats);
  hipLaunchKernelGGL(log_softmax50, dim3((BB * TT + 255) / 256), dim3(256), 0,
                     stream, feats);
  hipLaunchKernelGGL(crf_loss, dim3(1), dim3(1024), 0, stream, feats, target,
                     seq_len, trans, start_s, end_s, (float*)d_out);
}
